// CascadedAttention_78048145703166
// MI455X (gfx1250) — matrix operand order repaired and checked by reading, not compiled, not run on hardware
//
#include <hip/hip_runtime.h>
#include <math.h>

#define B_ 64
#define T_ 512
#define D_ 1024
#define U_ 28
#define G3 (3*U_)   // 84

typedef __attribute__((ext_vector_type(16))) __bf16 v16bf;
typedef __attribute__((ext_vector_type(8)))  float  v8f;
typedef unsigned short u16;
typedef unsigned int   u32;

__device__ __forceinline__ void sched_fence(){
#if __has_builtin(__builtin_amdgcn_sched_barrier)
  __builtin_amdgcn_sched_barrier(0);   // nothing crosses this point
#endif
}

// ---------- bf16 helpers (bf16 <-> f32 is a 16-bit shift; store uses RNE) ----------
__device__ __forceinline__ float bf2f(u16 v){
  union { u32 u; float f; } c; c.u = ((u32)v) << 16; return c.f;
}
__device__ __forceinline__ u16 f2bf(float f){
  union { u32 u; float f; } c; c.f = f;
  u32 r = c.u + 0x7FFFu + ((c.u >> 16) & 1u);   // round-to-nearest-even
  return (u16)(r >> 16);
}
// gfx1250 has V_TANH_F32 (trans unit) — use it when the builtin is declared.
__device__ __forceinline__ float fast_tanh(float x){
#if __has_builtin(__builtin_amdgcn_tanhf)
  return __builtin_amdgcn_tanhf(x);
#else
  // tanh(x) = 1 - 2/(1+exp(2x)); correct at both saturation ends, no sign fixup.
  return 1.0f - 2.0f / (1.0f + __expf(2.0f * x));
#endif
}
__device__ __forceinline__ float sigmoidf_(float x){
  return 1.0f / (1.0f + __expf(-x));
}

// ---------- one-shot: fp32 -> bf16 convert (x, 33.5M elems) ----------
__global__ void k_cvt_x(const float* __restrict__ x, u16* __restrict__ xbf, int n4){
  int i = blockIdx.x * blockDim.x + threadIdx.x;
  if (i < n4){
    float4 v = ((const float4*)x)[i];
    ushort4 o;
    o.x = f2bf(v.x); o.y = f2bf(v.y); o.z = f2bf(v.z); o.w = f2bf(v.w);
    ((ushort4*)xbf)[i] = o;
  }
}

// ---------- one-shot: Ua[k][n] -> UaT[n][k] in bf16 (so B-fragments are contiguous) ----------
__global__ void k_transpose_ua(const float* __restrict__ Ua, u16* __restrict__ UaT){
  __shared__ u16 tile[32][33];
  int n0 = blockIdx.x * 32, k0 = blockIdx.y * 32;
  int tx = threadIdx.x, ty = threadIdx.y;      // 32 x 8
  for (int r = 0; r < 32; r += 8)
    tile[ty + r][tx] = f2bf(Ua[(size_t)(k0 + ty + r) * D_ + n0 + tx]);
  __syncthreads();
  for (int r = 0; r < 32; r += 8)
    UaT[(size_t)(n0 + ty + r) * D_ + k0 + tx] = tile[tx][ty + r];
}

// ---------- one-shot GEMM: UaH = x @ Ua + ba1  (bf16 WMMA, fp32 accum, bf16 out) ----------
// Each wave computes a 16(M) x 128(N) strip. Per K-step: preload A fragment + ALL 8 B
// fragments (18 x b128 issued together, all in flight), sched_barrier pins the order so
// the scheduler cannot sink loads back into the WMMA burst (which would force register
// reuse and a full s_wait_loadcnt before every WMMA), then 8 back-to-back v_wmma.
// Fragment layouts per ISA 7.12.2 (16-bit A 16x32 / B 32x16, lane halves 0-15/16-31).
__global__ __launch_bounds__(128)
void k_gemm_uah(const u16* __restrict__ Abf,   // [32768][1024] bf16 (x flattened)
                const u16* __restrict__ BT,    // [1024][1024] bf16 = Ua^T
                const float* __restrict__ ba1, // [1024]
                u16* __restrict__ Cbf)         // [32768][1024] bf16
{
  int wave = blockIdx.x * (blockDim.x >> 5) + (threadIdx.x >> 5);
  int lane = threadIdx.x & 31;
  int mt = wave >> 3;                 // 0..2047  (M tile of 16)
  int ns = wave & 7;                  // 0..7     (N strip of 128)
  int m0 = mt * 16;
  int n0 = ns * 128;
  int half = lane >> 4;               // 0: lanes 0-15, 1: lanes 16-31
  int l16  = lane & 15;

  v8f acc[8] = {};

  const u16* arow = Abf + (size_t)(m0 + l16) * D_;
  const int aoff0 = half ? 8  : 0;    // A: K 0-7 / 8-15
  const int aoff1 = half ? 24 : 16;   // A: K 16-23 / 24-31
  const int boff  = half ? 8 : 0;     // B: K 0-7,16-23 / 8-15,24-31 (same order as A)
  const u16* bbase = BT + (size_t)(n0 + l16) * D_ + boff;  // + j*16*D_ per tile

#pragma unroll 1
  for (int k0 = 0; k0 < D_; k0 += 32){
    v16bf a;
    ((uint4*)&a)[0] = *(const uint4*)(arow + k0 + aoff0);
    ((uint4*)&a)[1] = *(const uint4*)(arow + k0 + aoff1);
    v16bf b[8];
#pragma unroll
    for (int j = 0; j < 8; ++j){
      const u16* brow = bbase + (size_t)(16 * j) * D_ + k0;
      ((uint4*)&b[j])[0] = *(const uint4*)(brow);
      ((uint4*)&b[j])[1] = *(const uint4*)(brow + 16);
    }
    sched_fence();                    // keep all 18 loads issued before any WMMA
#pragma unroll
    for (int j = 0; j < 8; ++j)
      acc[j] = __builtin_amdgcn_wmma_f32_16x16x32_bf16(
                 false, a, false, b[j], (short)0, acc[j], false, false);
    sched_fence();                    // keep next-iter loads out of the WMMA burst
  }
  // C layout: VGPR r, lane -> M = r + 8*half, N = 16*j + l16
#pragma unroll
  for (int j = 0; j < 8; ++j){
    int col = n0 + 16*j + l16;
    float bias = ba1[col];
#pragma unroll
    for (int r = 0; r < 8; ++r){
      int row = m0 + r + 8*half;
      Cbf[(size_t)row * D_ + col] = f2bf(acc[j][r] + bias);
    }
  }
}

// ---------- init: h0 = 0, so WaS(step0) = h0 @ Wa = 0 ----------
__global__ void k_init(float* __restrict__ WaS, float* __restrict__ h){
  int i = blockIdx.x * blockDim.x + threadIdx.x;
  if (i < B_ * D_) WaS[i] = 0.f;
  if (i < B_ * U_) h[i]   = 0.f;
}

// ---------- per step: scores[b,t] = sum_d tanh(UaH[b,t,d] + WaS[b,d]) * Va[d] + ba2 ----------
// One wave per t; WaS row + Va staged in LDS; UaH streamed as packed bf16 pairs (L2 resident).
__global__ __launch_bounds__(256)
void k_scores(const u16* __restrict__ UaH, const float* __restrict__ WaS,
              const float* __restrict__ Va, const float* __restrict__ ba2,
              float* __restrict__ scores)
{
  __shared__ float sW[D_];
  __shared__ float sV[D_];
  int b = blockIdx.y;
  int tid = threadIdx.x;
  for (int d = tid; d < D_; d += 256){
    sW[d] = WaS[b * D_ + d];
    sV[d] = Va[d];
  }
  __syncthreads();
  int wv = tid >> 5, lane = tid & 31;
  int t = blockIdx.x * 8 + wv;
  const u32* row32 = (const u32*)(UaH + (size_t)(b * T_ + t) * D_);
  float sum = 0.f;
  for (int p = lane; p < D_ / 2; p += 32){
    u32 v = row32[p];
    int d = 2 * p;
    sum += fast_tanh(bf2f((u16)v)         + sW[d])     * sV[d];
    sum += fast_tanh(bf2f((u16)(v >> 16)) + sW[d + 1]) * sV[d + 1];
  }
#pragma unroll
  for (int off = 16; off; off >>= 1) sum += __shfl_xor(sum, off, 32);
  if (lane == 0) scores[b * T_ + t] = sum + ba2[0];
}

// ---------- per step: softmax(scores[b,:]) fused with ctx[b,d] = sum_t attn[t]*x[b,t,d] ----------
// Each thread owns 2 adjacent d's (one packed u32 bf16 load per t).
__global__ __launch_bounds__(256)
void k_ctx(const u16* __restrict__ xbf, const float* __restrict__ scores,
           float* __restrict__ ctx)
{
  __shared__ float sa[T_];
  __shared__ float red[256];
  int b = blockIdx.y;
  int tid = threadIdx.x;
  float s0 = scores[b * T_ + tid];
  float s1 = scores[b * T_ + 256 + tid];
  red[tid] = fmaxf(s0, s1);
  __syncthreads();
  for (int off = 128; off; off >>= 1){
    if (tid < off) red[tid] = fmaxf(red[tid], red[tid + off]);
    __syncthreads();
  }
  float m = red[0];
  __syncthreads();
  float e0 = __expf(s0 - m), e1 = __expf(s1 - m);
  red[tid] = e0 + e1;
  __syncthreads();
  for (int off = 128; off; off >>= 1){
    if (tid < off) red[tid] += red[tid + off];
    __syncthreads();
  }
  float inv = 1.0f / red[0];
  sa[tid]       = e0 * inv;
  sa[tid + 256] = e1 * inv;
  __syncthreads();

  int d0 = blockIdx.x * 512 + tid * 2;
  const u32* xp = (const u32*)(xbf + (size_t)b * T_ * D_ + d0);  // coalesced, 2 bf16/thread
  float acc0 = 0.f, acc1 = 0.f;
#pragma unroll 4
  for (int t = 0; t < T_; ++t){
    u32 v = xp[(size_t)t * (D_ / 2)];
    float w = sa[t];
    acc0 += w * bf2f((u16)v);
    acc1 += w * bf2f((u16)(v >> 16));
  }
  ctx[b * D_ + d0]     = acc0;
  ctx[b * D_ + d0 + 1] = acc1;
}

// ---------- per step: GRU cell (reset_after=True) + write ys[b,t,:] + next WaS = h_new @ Wa ----------
__global__ __launch_bounds__(128)
void k_gru(const float* __restrict__ ctx, const float* __restrict__ gk,
           const float* __restrict__ grk, const float* __restrict__ gbias,
           const float* __restrict__ Wa, float* __restrict__ h,
           float* __restrict__ WaS, float* __restrict__ out, int t)
{
  __shared__ float sc[D_];
  __shared__ float sh[U_];
  __shared__ float sg[2 * G3];
  __shared__ float shn[U_];
  int b = blockIdx.x;
  int tid = threadIdx.x;
  for (int d = tid; d < D_; d += 128) sc[d] = ctx[b * D_ + d];
  if (tid < U_) sh[tid] = h[b * U_ + tid];
  __syncthreads();
  if (tid < G3){
    float xg = gbias[tid];
    for (int d = 0; d < D_; ++d) xg += sc[d] * gk[d * G3 + tid];   // coalesced across tid
    float hg = gbias[G3 + tid];
#pragma unroll
    for (int u = 0; u < U_; ++u) hg += sh[u] * grk[u * G3 + tid];
    sg[tid] = xg; sg[G3 + tid] = hg;
  }
  __syncthreads();
  if (tid < U_){
    float z  = sigmoidf_(sg[tid]          + sg[G3 + tid]);
    float r  = sigmoidf_(sg[U_ + tid]     + sg[G3 + U_ + tid]);
    float hc = fast_tanh(sg[2*U_ + tid] + r * sg[G3 + 2*U_ + tid]);
    float hn = z * sh[tid] + (1.0f - z) * hc;
    shn[tid] = hn;
    h[b * U_ + tid] = hn;
    out[((size_t)b * T_ + t) * U_ + tid] = hn;
  }
  __syncthreads();
  for (int d = tid; d < D_; d += 128){
    float s = 0.f;
#pragma unroll
    for (int u = 0; u < U_; ++u) s += shn[u] * Wa[u * D_ + d];     // Wa: [U,D], coalesced in d
    WaS[b * D_ + d] = s;
  }
}

extern "C" void kernel_launch(void* const* d_in, const int* in_sizes, int n_in,
                              void* d_out, int out_size, void* d_ws, size_t ws_size,
                              hipStream_t stream)
{
  const float* x   = (const float*)d_in[0];
  const float* Wa  = (const float*)d_in[1];
  const float* Ua  = (const float*)d_in[2];
  const float* Va  = (const float*)d_in[3];
  const float* ba1 = (const float*)d_in[4];
  const float* ba2 = (const float*)d_in[5];
  const float* gk  = (const float*)d_in[6];
  const float* grk = (const float*)d_in[7];
  const float* gb  = (const float*)d_in[8];
  float* out = (float*)d_out;
  (void)in_sizes; (void)n_in; (void)out_size; (void)ws_size;

  char* ws = (char*)d_ws;
  size_t off = 0;
  auto alloc = [&](size_t bytes) -> char* {
    char* p = ws + off;
    off = (off + bytes + 255) & ~(size_t)255;
    return p;
  };
  u16*   xbf    = (u16*)  alloc((size_t)B_ * T_ * D_ * 2);   // 67 MB, L2-resident
  u16*   UaH    = (u16*)  alloc((size_t)B_ * T_ * D_ * 2);   // 67 MB, L2-resident
  u16*   UaT    = (u16*)  alloc((size_t)D_ * D_ * 2);        //  2 MB
  float* WaS    = (float*)alloc((size_t)B_ * D_ * 4);
  float* scores = (float*)alloc((size_t)B_ * T_ * 4);
  float* ctx    = (float*)alloc((size_t)B_ * D_ * 4);
  float* h      = (float*)alloc((size_t)B_ * U_ * 4);

  const int n4 = B_ * T_ * D_ / 4;
  k_cvt_x<<<n4 / 256, 256, 0, stream>>>(x, xbf, n4);
  k_transpose_ua<<<dim3(D_ / 32, D_ / 32), dim3(32, 8), 0, stream>>>(Ua, UaT);
  // 2048 M-tiles x 8 N-strips = 16384 waves; 4 waves/block
  k_gemm_uah<<<4096, 128, 0, stream>>>(xbf, UaT, ba1, UaH);
  k_init<<<(B_ * D_ + 255) / 256, 256, 0, stream>>>(WaS, h);

  for (int t = 0; t < T_; ++t){
    k_scores<<<dim3(T_ / 8, B_), 256, 0, stream>>>(UaH, WaS, Va, ba2, scores);
    k_ctx<<<dim3(D_ / 512, B_), 256, 0, stream>>>(xbf, scores, ctx);
    k_gru<<<B_, 128, 0, stream>>>(ctx, gk, grk, gb, Wa, h, WaS, out, t);
  }
}
